// CausalSelfAttention_78228534329873
// MI455X (gfx1250) — matrix operand order repaired and checked by reading, not compiled, not run on hardware
//
#include <hip/hip_runtime.h>

// ---------------------------------------------------------------------------
// CDNA5 (gfx1250) causal self-attention, bf16 WMMA path, f32 accumulate.
// B=4, T=2048, D=1024, H=16, DH=64.
// GEMMs: 32x64 tile per wave, A staged to LDS via Tensor Data Mover (TDM),
// double-buffered (tensor_load_to_lds + s_wait_tensorcnt), B streamed from L2.
// ---------------------------------------------------------------------------

typedef __bf16 v16bf __attribute__((ext_vector_type(16)));
typedef __bf16 v8bf  __attribute__((ext_vector_type(8)));
typedef float  v8f   __attribute__((ext_vector_type(8)));
typedef unsigned int u32x4 __attribute__((ext_vector_type(4)));
typedef int          i32x4 __attribute__((ext_vector_type(4)));
typedef int          i32x8 __attribute__((ext_vector_type(8)));

constexpr int NB  = 4;
constexpr int NT  = 2048;
constexpr int ND  = 1024;
constexpr int NH  = 16;
constexpr int NDH = 64;
constexpr int BT  = NB * NT;      // 8192 rows
constexpr int ND3 = 3 * ND;       // 3072
constexpr int KC  = 256;          // K-chunk staged in LDS (32 x 256 bf16 = 16 KB)

#if __has_builtin(__builtin_amdgcn_tensor_load_to_lds)
#define HAVE_TDM 1
#else
#define HAVE_TDM 0
#endif

static __device__ __forceinline__ v8f wmma_bf16(v16bf a, v16bf b, v8f c) {
    // D = A(16x32 bf16) * B(32x16 bf16) + C(16x16 f32)
    return __builtin_amdgcn_wmma_f32_16x16x32_bf16(
        /*neg_a=*/false, a, /*neg_b=*/false, b,
        /*c_mod=*/(short)0, c, /*reuse_a=*/false, /*reuse_b=*/false);
}

// A-fragment (16x32, 16-bit): lane holds row M=lane%16,
// K = (lane/16)*16 + {0..15}  -> two 16-byte chunks.
static __device__ __forceinline__ v16bf load_a32(const __bf16* rowp, int hi) {
    union { v16bf v; v8bf h[2]; } u;
    u.h[0] = *(const v8bf*)(rowp + hi * 16);
    u.h[1] = *(const v8bf*)(rowp + hi * 16 + 8);
    return u.v;
}

static __device__ __forceinline__ void wait_tensor0() {
#if __has_builtin(__builtin_amdgcn_s_wait_tensorcnt)
    __builtin_amdgcn_s_wait_tensorcnt(0);
#else
    asm volatile("s_wait_tensorcnt 0" ::: "memory");
#endif
}

#if HAVE_TDM
// Issue TENSOR_LOAD_TO_LDS of a 2D bf16 tile: tdim1 rows x tdim0 cols,
// row stride `stride_elems` in global, packed contiguously at lds_off.
// D# layout per CDNA5 ISA ch.8 (group0: count/lds/global/type, group1: dims).
static __device__ __forceinline__ void tdm_load_2d_bf16(const void* gptr, unsigned lds_off,
                                                        unsigned tdim0, unsigned tdim1,
                                                        unsigned long long stride_elems) {
    unsigned long long ga = (unsigned long long)gptr;
    u32x4 g0;
    g0[0] = 1u;                                             // count=1 (valid), user mode
    g0[1] = lds_off;                                        // lds_addr (bytes)
    g0[2] = (unsigned)(ga & 0xffffffffu);                   // global_addr[31:0]
    g0[3] = (unsigned)((ga >> 32) & 0x01ffffffu)            // global_addr[56:32]
          | (2u << 30);                                     // type=2 ("image")
    i32x8 g1;
    g1[0] = (int)(1u << 16);                                // data_size=1 -> 2 bytes
    g1[1] = (int)((tdim0 & 0xffffu) << 16);                 // tensor_dim0[15:0]
    g1[2] = (int)((tdim0 >> 16) | ((tdim1 & 0xffffu) << 16)); // dim0[31:16] | dim1[15:0]
    g1[3] = (int)((tdim1 >> 16) | ((tdim0 & 0xffffu) << 16)); // dim1[31:16] | tile_dim0
    g1[4] = (int)(tdim1 & 0xffffu);                         // tile_dim1 ; tile_dim2=0
    g1[5] = (int)(stride_elems & 0xffffffffull);            // tensor_dim0_stride[31:0]
    g1[6] = (int)(stride_elems >> 32);                      // stride[47:32]; dim1_stride=0
    g1[7] = 0;
    i32x4 z4 = {0, 0, 0, 0};
#if defined(__clang_major__) && __clang_major__ >= 23
    i32x8 z8 = {0, 0, 0, 0, 0, 0, 0, 0};
    __builtin_amdgcn_tensor_load_to_lds(g0, g1, z4, z4, z8, 0);
#else
    __builtin_amdgcn_tensor_load_to_lds(g0, g1, z4, z4, 0);
#endif
}
#endif

// Fallback A-stage: cooperative copy of 32 x KC bf16 (all 128 threads).
static __device__ __forceinline__ void coop_copy_A(const __bf16* __restrict__ src,
                                                   __bf16* __restrict__ dst, int tid) {
#pragma unroll
    for (int i = 0; i < (32 * KC) / (8 * 128); ++i) {
        int idx = i * 128 + tid;            // 16-byte chunk index
        int row = idx >> 5;                 // KC/8 = 32 chunks per row
        int c8  = idx & 31;
        *(v8bf*)(dst + row * KC + c8 * 8) = *(const v8bf*)(src + (size_t)row * ND + c8 * 8);
    }
}

// Shared inner loop: one LDS-resident A chunk (32 x KC), stream B, 8 WMMA per K-step.
static __device__ __forceinline__ void gemm_chunk(const __bf16* A, const __bf16* __restrict__ wT,
                                                  int n0, int gk0, int ln, int hi,
                                                  v8f acc[2][4]) {
#pragma unroll
    for (int kk = 0; kk < KC; kk += 32) {
        v16bf a0 = load_a32(A + ln * KC + kk, hi);          // rows 0..15
        v16bf a1 = load_a32(A + (16 + ln) * KC + kk, hi);   // rows 16..31
#pragma unroll
        for (int nt = 0; nt < 4; ++nt) {
            const __bf16* bp = wT + (size_t)(n0 + nt * 16 + ln) * ND + gk0 + kk + hi * 16;
            v16bf b = *(const v16bf*)bp;                    // B-frag: col N=ln, K contiguous
            acc[0][nt] = wmma_bf16(a0, b, acc[0][nt]);
            acc[1][nt] = wmma_bf16(a1, b, acc[1][nt]);
        }
    }
}

// ---------------------------------------------------------------------------
// Conversion / transpose kernels (bandwidth-trivial: ~25 MB total)
// ---------------------------------------------------------------------------
__global__ __launch_bounds__(256) void cvt_f32_bf16(const float* __restrict__ in,
                                                    __bf16* __restrict__ out, int n) {
    int i = blockIdx.x * 256 + threadIdx.x;
    if (i < n) out[i] = (__bf16)in[i];
}

// out[e*rows_in + d] = bf16(in[d*cols_in + e])
__global__ __launch_bounds__(256) void cvt_transpose_bf16(const float* __restrict__ in,
                                                          __bf16* __restrict__ out,
                                                          int rows_in, int cols_in) {
    int i = blockIdx.x * 256 + threadIdx.x;
    int n = rows_in * cols_in;
    if (i >= n) return;
    int e = i / rows_in;
    int d = i - e * rows_in;
    out[(size_t)e * rows_in + d] = (__bf16)in[(size_t)d * cols_in + e];
}

// ---------------------------------------------------------------------------
// QKV projection: qkv[t,e] = sum_d xb[t,d] * wqkvT[e,d]
// Block: 32 rows, 4 waves x 64 cols. A tile TDM-staged to LDS, double-buffered.
// Scatter into q[B,H,T,DH], k[B,H,T,DH], vT[B,H,DH,T].
// ---------------------------------------------------------------------------
__global__ __launch_bounds__(128) void qkv_gemm_kernel(const __bf16* __restrict__ xb,
                                                       const __bf16* __restrict__ wT,
                                                       __bf16* __restrict__ q,
                                                       __bf16* __restrict__ k,
                                                       __bf16* __restrict__ vT) {
    __shared__ __align__(16) __bf16 Abuf[2][32 * KC];      // 2 x 16 KB
    const int tid  = threadIdx.x;
    const int lane = tid & 31;
    const int wid  = tid >> 5;
    const int ln   = lane & 15;
    const int hi   = lane >> 4;
    const int m0   = blockIdx.x * 32;
    const int n0   = (blockIdx.y * 4 + wid) * 64;

    const __bf16* Asrc = xb + (size_t)m0 * ND;
    v8f acc[2][4] = {};

#if HAVE_TDM
    if (wid == 0) {
        tdm_load_2d_bf16(Asrc, (unsigned)(size_t)&Abuf[0][0], KC, 32, ND);
        wait_tensor0();
    }
    __syncthreads();
    int buf = 0;
    for (int c = 0; c < ND / KC; ++c) {
        if (wid == 0 && c + 1 < ND / KC)
            tdm_load_2d_bf16(Asrc + (c + 1) * KC, (unsigned)(size_t)&Abuf[buf ^ 1][0],
                             KC, 32, ND);
        gemm_chunk(&Abuf[buf][0], wT, n0, c * KC, ln, hi, acc);
        if (wid == 0 && c + 1 < ND / KC) wait_tensor0();
        __syncthreads();
        buf ^= 1;
    }
#else
    for (int c = 0; c < ND / KC; ++c) {
        coop_copy_A(Asrc + c * KC, &Abuf[0][0], tid);
        __syncthreads();
        gemm_chunk(&Abuf[0][0], wT, n0, c * KC, ln, hi, acc);
        __syncthreads();
    }
#endif

    // n0 is a multiple of 64 -> whole slab lies in one of {q,k,v} and one head
    const int sect = n0 / ND;                               // 0=q 1=k 2=v
    const int h    = (n0 % ND) / NDH;
#pragma unroll
    for (int mt = 0; mt < 2; ++mt) {
#pragma unroll
        for (int nt = 0; nt < 4; ++nt) {
            const int dh = nt * 16 + ln;
#pragma unroll
            for (int r = 0; r < 8; ++r) {
                const int t   = m0 + mt * 16 + r + 8 * hi;
                const int b   = t >> 11;                    // /NT
                const int tin = t & (NT - 1);
                const __bf16 val = (__bf16)acc[mt][nt][r];
                const size_t bh = (size_t)(b * NH + h);
                if (sect == 0)      q[(bh * NT + tin) * NDH + dh]  = val;
                else if (sect == 1) k[(bh * NT + tin) * NDH + dh]  = val;
                else                vT[(bh * NDH + dh) * NT + tin] = val;
            }
        }
    }
}

// ---------------------------------------------------------------------------
// Flash-style causal attention. One wave = one 16-row q tile of one (b,h).
// Key loop in 32-key chunks (2 S-tiles), online softmax, PV via WMMA.
// ---------------------------------------------------------------------------
__global__ __launch_bounds__(128) void attn_kernel(const __bf16* __restrict__ q,
                                                   const __bf16* __restrict__ k,
                                                   const __bf16* __restrict__ vT,
                                                   __bf16* __restrict__ y) {
    __shared__ __align__(16) __bf16 ldsP[4][16 * 32];       // per-wave P staging

    const int lane = threadIdx.x & 31;
    const int wid  = threadIdx.x >> 5;
    const int ln   = lane & 15;
    const int hi   = lane >> 4;
    const int bh   = blockIdx.x;                            // 0..63
    const int b    = bh >> 4;
    const int h    = bh & 15;
    const int qt0  = (blockIdx.y * 4 + wid) * 16;           // q-tile row base

    const __bf16* qrow = q + ((size_t)bh * NT + qt0 + ln) * NDH;
    const v16bf aq0 = load_a32(qrow, hi);                   // K-slice [0,32)
    const v16bf aq1 = load_a32(qrow + 32, hi);              // K-slice [32,64)

    const __bf16* kbase = k  + (size_t)bh * NT * NDH;
    const __bf16* vbase = vT + (size_t)bh * NDH * NT;

    v8f o[4] = {};
    float m[8], l[8];
#pragma unroll
    for (int r = 0; r < 8; ++r) { m[r] = -1e30f; l[r] = 0.0f; }

    __bf16* P = ldsP[wid];

    for (int kt0 = 0; kt0 < qt0 + 16; kt0 += 32) {
        // ---- S = (Q K^T) * 1/sqrt(DH), two 16-col tiles ----
        v8f s[2] = {};
#pragma unroll
        for (int c = 0; c < 2; ++c) {
            const int key = kt0 + c * 16 + ln;
            const __bf16* kp = kbase + (size_t)key * NDH + hi * 16;
            v16bf b0 = *(const v16bf*)kp;                   // K in [0,32)
            v16bf b1 = *(const v16bf*)(kp + 32);            // K in [32,64)
            s[c] = wmma_bf16(aq0, b0, s[c]);
            s[c] = wmma_bf16(aq1, b1, s[c]);
        }

        const bool needmask = (kt0 + 31 > qt0);
        float p0[8], p1[8];
#pragma unroll
        for (int r = 0; r < 8; ++r) {
            float v0 = s[0][r] * 0.125f;                    // 1/sqrt(64)
            float v1 = s[1][r] * 0.125f;
            if (needmask) {
                const int qq = qt0 + r + 8 * hi;
                if (kt0 + ln > qq)      v0 = -1e30f;
                if (kt0 + 16 + ln > qq) v1 = -1e30f;
            }
            p0[r] = v0; p1[r] = v1;
        }

        // ---- row max over 32 keys (xor-shuffle within 16-lane column group) ----
        float rm[8];
#pragma unroll
        for (int r = 0; r < 8; ++r) rm[r] = fmaxf(p0[r], p1[r]);
#pragma unroll
        for (int off = 1; off < 16; off <<= 1)
#pragma unroll
            for (int r = 0; r < 8; ++r)
                rm[r] = fmaxf(rm[r], __shfl_xor(rm[r], off, 32));

        float alpha[8];
#pragma unroll
        for (int r = 0; r < 8; ++r) {
            const float mn = fmaxf(m[r], rm[r]);
            alpha[r] = __expf(m[r] - mn);
            m[r] = mn;
        }

        float rs[8];
#pragma unroll
        for (int r = 0; r < 8; ++r) {
            p0[r] = __expf(p0[r] - m[r]);
            p1[r] = __expf(p1[r] - m[r]);
            rs[r] = p0[r] + p1[r];
        }
#pragma unroll
        for (int off = 1; off < 16; off <<= 1)
#pragma unroll
            for (int r = 0; r < 8; ++r)
                rs[r] += __shfl_xor(rs[r], off, 32);

#pragma unroll
        for (int r = 0; r < 8; ++r) {
            l[r] = l[r] * alpha[r] + rs[r];
#pragma unroll
            for (int nt = 0; nt < 4; ++nt) o[nt][r] *= alpha[r];
        }

        // ---- C-layout -> A-fragment layout for P via per-wave LDS ----
#pragma unroll
        for (int r = 0; r < 8; ++r) {
            const int row = r + 8 * hi;                     // C-layout row
            P[row * 32 + ln]      = (__bf16)p0[r];
            P[row * 32 + 16 + ln] = (__bf16)p1[r];
        }
        asm volatile("s_wait_dscnt 0" ::: "memory");        // in-wave LDS RAW ordering
        const v16bf ap = load_a32(P + ln * 32, hi);         // A-frag of P[16x32]

        // ---- O += P * V  (4 dh tiles, K = 32 keys) ----
#pragma unroll
        for (int nt = 0; nt < 4; ++nt) {
            const int d0 = nt * 16 + ln;
            const __bf16* vp = vbase + (size_t)d0 * NT + kt0 + hi * 16;
            v16bf bv = *(const v16bf*)vp;                   // contiguous over keys
            o[nt] = wmma_bf16(ap, bv, o[nt]);
        }
    }

    // ---- epilogue: O /= l, write y[B,T,D] as bf16 ----
#pragma unroll
    for (int r = 0; r < 8; ++r) {
        const float inv = 1.0f / l[r];
        const int row = qt0 + r + 8 * hi;
        __bf16* yp = y + ((size_t)b * NT + row) * ND + h * NDH;
#pragma unroll
        for (int nt = 0; nt < 4; ++nt)
            yp[nt * 16 + ln] = (__bf16)(o[nt][r] * inv);
    }
}

// ---------------------------------------------------------------------------
// Output projection: out[t,d] = sum_e y[t,e] * woutT[d,e], fp32 output.
// Same 32x64 tiling with TDM-staged A.
// ---------------------------------------------------------------------------
__global__ __launch_bounds__(128) void out_gemm_kernel(const __bf16* __restrict__ yb,
                                                       const __bf16* __restrict__ wT,
                                                       float* __restrict__ out) {
    __shared__ __align__(16) __bf16 Abuf[2][32 * KC];
    const int tid  = threadIdx.x;
    const int lane = tid & 31;
    const int wid  = tid >> 5;
    const int ln   = lane & 15;
    const int hi   = lane >> 4;
    const int m0   = blockIdx.x * 32;
    const int n0   = (blockIdx.y * 4 + wid) * 64;

    const __bf16* Asrc = yb + (size_t)m0 * ND;
    v8f acc[2][4] = {};

#if HAVE_TDM
    if (wid == 0) {
        tdm_load_2d_bf16(Asrc, (unsigned)(size_t)&Abuf[0][0], KC, 32, ND);
        wait_tensor0();
    }
    __syncthreads();
    int buf = 0;
    for (int c = 0; c < ND / KC; ++c) {
        if (wid == 0 && c + 1 < ND / KC)
            tdm_load_2d_bf16(Asrc + (c + 1) * KC, (unsigned)(size_t)&Abuf[buf ^ 1][0],
                             KC, 32, ND);
        gemm_chunk(&Abuf[buf][0], wT, n0, c * KC, ln, hi, acc);
        if (wid == 0 && c + 1 < ND / KC) wait_tensor0();
        __syncthreads();
        buf ^= 1;
    }
#else
    for (int c = 0; c < ND / KC; ++c) {
        coop_copy_A(Asrc + c * KC, &Abuf[0][0], tid);
        __syncthreads();
        gemm_chunk(&Abuf[0][0], wT, n0, c * KC, ln, hi, acc);
        __syncthreads();
    }
#endif

#pragma unroll
    for (int mt = 0; mt < 2; ++mt)
#pragma unroll
        for (int nt = 0; nt < 4; ++nt)
#pragma unroll
            for (int r = 0; r < 8; ++r)
                out[(size_t)(m0 + mt * 16 + r + 8 * hi) * ND + n0 + nt * 16 + ln]
                    = acc[mt][nt][r];
}

// ---------------------------------------------------------------------------
// Host-side launch. Workspace layout (bytes):
//   [0)         xb   : 16 MB (reused as y after attention)
//   [16777216)  wqkvT:  6 MB
//   [23068672)  woutT:  2 MB
//   [25165824)  q    : 16 MB
//   [41943040)  k    : 16 MB
//   [58720256)  vT   : 16 MB   -> total 72 MB
// ---------------------------------------------------------------------------
extern "C" void kernel_launch(void* const* d_in, const int* in_sizes, int n_in,
                              void* d_out, int out_size, void* d_ws, size_t ws_size,
                              hipStream_t stream) {
    (void)in_sizes; (void)n_in; (void)out_size; (void)ws_size;

    const float* x     = (const float*)d_in[0];   // [B,T,D]
    const float* w_qkv = (const float*)d_in[1];   // [D,3D]
    const float* w_out = (const float*)d_in[2];   // [D,D]
    float*       out   = (float*)d_out;           // [B,T,D]

    char* ws = (char*)d_ws;
    __bf16* xb    = (__bf16*)(ws + 0);
    __bf16* wqkvT = (__bf16*)(ws + 16777216);
    __bf16* woutT = (__bf16*)(ws + 23068672);
    __bf16* qb    = (__bf16*)(ws + 25165824);
    __bf16* kb    = (__bf16*)(ws + 41943040);
    __bf16* vTb   = (__bf16*)(ws + 58720256);
    __bf16* yb    = xb;                            // reuse (x dead after QKV gemm)

    {
        int n = BT * ND;
        cvt_f32_bf16<<<(n + 255) / 256, 256, 0, stream>>>(x, xb, n);
    }
    {
        int n = ND3 * ND;   // wqkvT[e,d] = w_qkv[d,e]
        cvt_transpose_bf16<<<(n + 255) / 256, 256, 0, stream>>>(w_qkv, wqkvT, ND, ND3);
    }
    {
        int n = ND * ND;
        cvt_transpose_bf16<<<(n + 255) / 256, 256, 0, stream>>>(w_out, woutT, ND, ND);
    }

    // QKV projection: M tiles = 8192/32 = 256; N slabs = 3072/64 = 48 -> grid.y = 12
    qkv_gemm_kernel<<<dim3(BT / 32, (ND3 / 64) / 4), 128, 0, stream>>>(xb, wqkvT, qb, kb, vTb);

    // attention: 64 (b,h) x 128 q-tiles / 4 waves per block
    attn_kernel<<<dim3(NB * NH, (NT / 16) / 4), 128, 0, stream>>>(qb, kb, vTb, yb);

    // output projection: N slabs = 1024/64 = 16 -> grid.y = 4
    out_gemm_kernel<<<dim3(BT / 32, (ND / 64) / 4), 128, 0, stream>>>(yb, woutT, out);
}
